// MultiAgentDRQN_60851096650253
// MI455X (gfx1250) — compile-verified, matrix operand order checked by reading, not run on hardware
//
#include <hip/hip_runtime.h>

// ---------------------------------------------------------------------------
// Multi-agent DRQN forward on gfx1250 (MI455X).
// - one wave32 per 16-row GRU tile, T-loop inside the wave
// - all GEMMs as v_wmma_f32_16x16x32_f16 (f16 in, f32 accumulate)
// - f16 weights resident in LDS (staged once; 62 KB of the 320 KB WGP LDS)
// - obs tiles double-buffered into LDS with global_load_async_to_lds_b128
//   (ASYNCcnt) so the HBM stream for step t+1 overlaps step t's compute,
//   plus global_prefetch for t+2 toward L2.
// ---------------------------------------------------------------------------

typedef _Float16 half_t;
typedef __attribute__((ext_vector_type(4)))  _Float16 v4h;
typedef __attribute__((ext_vector_type(8)))  _Float16 v8h;
typedef __attribute__((ext_vector_type(16))) _Float16 v16h;
typedef __attribute__((ext_vector_type(8)))  float    v8f;

constexpr int kB = 256, kT = 256, kA = 8, kOBS = 64, kACT = 16, kH = 64;
constexpr int kIN = 88, kINP = 96;          // IN padded to 3 K-chunks of 32
constexpr int WAVES = 2;                    // waves (16-row tiles) per workgroup
// f16 weight layout (element offsets), both in workspace and in LDS
constexpr int OW1 = 0;                      // [64][96] (zero-padded cols 88..95)
constexpr int OWIH = OW1 + kH * kINP;       // 6144,  [192][64]
constexpr int OWHH = OWIH + 3 * kH * kH;    // 18432, [192][64]
constexpr int OW2 = OWHH + 3 * kH * kH;     // 30720, [16][64]
constexpr int WTOT = OW2 + kACT * kH;       // 31744 halves = 62 KB
// bias LDS layout (float offsets)
constexpr int OB1 = 0, OBIH = 64, OBHH = 256, OB2 = 448, BTOT = 464;
// output regions (element offsets inside d_out)
constexpr size_t MV_OFF = (size_t)kB * kT * kA * kACT;    // nn_output size
constexpr size_t MA_OFF = MV_OFF + (size_t)kB * kT * kA;  // + max_values size

// Same-wave LDS store -> cross-lane load ordering (no workgroup barrier).
#define WAVE_LDS_SYNC()  asm volatile("s_wait_dscnt 0x0" ::: "memory")
#define WAVE_ASYNC_WAIT() asm volatile("s_wait_asynccnt 0x0" ::: "memory")

// --------------------------- weight conversion -----------------------------
__global__ void drqn_cvt_weights(const float* __restrict__ W1,
                                 const float* __restrict__ Wih,
                                 const float* __restrict__ Whh,
                                 const float* __restrict__ W2,
                                 half_t* __restrict__ wh) {
  int i = blockIdx.x * blockDim.x + threadIdx.x;
  if (i >= WTOT) return;
  float v;
  if (i < OWIH) {                       // W1 [64][88] -> [64][96] zero-padded
    int r = i / kINP, c = i % kINP;
    v = (c < kIN) ? W1[r * kIN + c] : 0.0f;
  } else if (i < OWHH) {
    v = Wih[i - OWIH];
  } else if (i < OW2) {
    v = Whh[i - OWHH];
  } else {
    v = W2[i - OW2];
  }
  wh[i] = (half_t)v;
}

// --------------------------- fragment helpers ------------------------------
// 16-bit fragment K-pattern (ISA 7.12.2): lane<16 -> K = koff+{0..7,16..23},
// lane>=16 -> K = koff+{8..15,24..31}. A: lane carries row; B: lane carries col.
__device__ __forceinline__ v16h ld_frag16(const half_t* base, int pitch,
                                          int row, int lane, int koff) {
  int hi = (lane & 16) ? 8 : 0;
  const half_t* p = base + row * pitch + koff + hi;
  union { v16h v; v8h h[2]; } u;
  u.h[0] = *(const v8h*)(p);
  u.h[1] = *(const v8h*)(p + 16);
  return u.v;
}
__device__ __forceinline__ v16h ld_fragA(const half_t* base, int pitch,
                                         int lane, int koff) {
  return ld_frag16(base, pitch, lane & 15, lane, koff);
}
__device__ __forceinline__ v16h ld_fragB(const half_t* wbase, int pitch,
                                         int lane, int ncol0, int koff) {
  return ld_frag16(wbase, pitch, ncol0 + (lane & 15), lane, koff);
}

// Build an A-fragment (f16) straight from the f32 obs tile in LDS.
__device__ __forceinline__ v16h frag_from_obs(const float* base, int lane,
                                              int koff) {
  int row = lane & 15;
  int hi = (lane & 16) ? 8 : 0;
  const float* q = base + row * kOBS + koff + hi;
  float4 a0 = *(const float4*)(q);
  float4 a1 = *(const float4*)(q + 4);
  float4 b0 = *(const float4*)(q + 16);
  float4 b1 = *(const float4*)(q + 20);
  union { v16h v; v4h h4[4]; } u;
  u.h4[0] = (v4h){(half_t)a0.x, (half_t)a0.y, (half_t)a0.z, (half_t)a0.w};
  u.h4[1] = (v4h){(half_t)a1.x, (half_t)a1.y, (half_t)a1.z, (half_t)a1.w};
  u.h4[2] = (v4h){(half_t)b0.x, (half_t)b0.y, (half_t)b0.z, (half_t)b0.w};
  u.h4[3] = (v4h){(half_t)b1.x, (half_t)b1.y, (half_t)b1.z, (half_t)b1.w};
  return u.v;
}

// A-fragment for x columns 64..95: agent one-hot (8) | last-action one-hot
// (16) | zero pad (8). Computed arithmetically, no memory traffic.
__device__ __forceinline__ v16h frag_onehot(int lane, int aa, int actp) {
  int hi = (lane & 16) ? 8 : 0;
  union { v16h v; half_t e[16]; } u;
#pragma unroll
  for (int j = 0; j < 8; ++j) {
    int c0 = hi + j;        // column - 64, group 0 (0..15)
    int c1 = hi + 16 + j;   // column - 64, group 1 (16..31)
    float v0 = (c0 < 8) ? (c0 == aa ? 1.0f : 0.0f)
                        : ((c0 - 8) == actp ? 1.0f : 0.0f);
    float v1 = (c1 < 24) ? ((c1 - 8) == actp ? 1.0f : 0.0f) : 0.0f;
    u.e[j] = (half_t)v0;
    u.e[8 + j] = (half_t)v1;
  }
  return u.v;
}

__device__ __forceinline__ v8f wmma_f16(v16h a, v16h b, v8f c) {
  return __builtin_amdgcn_wmma_f32_16x16x32_f16(false, a, false, b,
                                                (short)0, c, false, false);
}

__device__ __forceinline__ v8f splat8(float x) {
  v8f r;
#pragma unroll
  for (int i = 0; i < 8; ++i) r[i] = x;
  return r;
}

__device__ __forceinline__ float sigmf(float x) {
  return 1.0f / (1.0f + __expf(-x));
}

// Async copy of one 16x64 f32 obs tile (4 KB) into LDS; tracked by ASYNCcnt.
__device__ __forceinline__ void issue_obs_async(const float* __restrict__ obs,
                                                int n0, int t,
                                                unsigned lds_base, int lane) {
#pragma unroll
  for (int i = 0; i < 8; ++i) {
    int flat = i * 32 + lane;               // 256 lanes-worth of b128
    int r = flat >> 4, c4 = (flat & 15) * 4;
    int n = n0 + r, bb = n >> 3, aa = n & 7;
    const float* g = obs + ((size_t)bb * kT * kA + (size_t)t * kA + aa) * kOBS + c4;
    unsigned la = lds_base + (unsigned)(r * kOBS + c4) * 4u;
    asm volatile("global_load_async_to_lds_b128 %0, %1, off"
                 :: "v"(la), "v"(g) : "memory");
  }
}

// ------------------------------ main kernel --------------------------------
__global__ __launch_bounds__(32 * WAVES) void drqn_forward(
    const float* __restrict__ obs,      // [B,T,A,OBS]
    const int*   __restrict__ act,      // [B,T,A,1]
    const float* __restrict__ b1, const float* __restrict__ bih,
    const float* __restrict__ bhh, const float* __restrict__ b2,
    const half_t* __restrict__ wh,      // f16 weights in workspace
    float* __restrict__ out) {
  __shared__ __align__(16) half_t s_w[WTOT];                // all weights, f16
  __shared__ float s_b[BTOT];                               // all biases, f32
  __shared__ __align__(16) float  s_obs[WAVES][2][16 * kOBS]; // async dbl buf
  __shared__ __align__(16) half_t s_x1[WAVES][16 * kH];     // relu(fc1) tile
  __shared__ __align__(16) half_t s_h[WAVES][2 * 16 * kH];  // h double buffer

  const int tid = threadIdx.x;
  // Stage weights (one uint4 = 8 halves per iteration) and biases into LDS.
  {
    const uint4* src = (const uint4*)wh;
    uint4* dst = (uint4*)s_w;
    for (int i = tid; i < WTOT / 8; i += blockDim.x) dst[i] = src[i];
    for (int i = tid; i < BTOT; i += blockDim.x) {
      float v;
      if (i < OBIH)      v = b1[i];
      else if (i < OBHH) v = bih[i - OBIH];
      else if (i < OB2)  v = bhh[i - OBHH];
      else               v = b2[i - OB2];
      s_b[i] = v;
    }
  }

  const int w = tid >> 5, lane = tid & 31;
  half_t* x1b = s_x1[w];
  half_t* hbb = s_h[w];
  for (int i = lane; i < 16 * kH; i += 32) hbb[i] = (half_t)0.0f;  // h0 = 0
  __syncthreads();

  const int tile = blockIdx.x * WAVES + w;
  const int n0 = tile * 16;                 // first batch row of this tile
  const int colc = lane & 15;               // C/D fragment column
  const int rbase = (lane & 16) ? 8 : 0;    // C/D fragment row base
  const int aa_r = (n0 + (lane & 15)) & 7;  // agent slot of this lane's row
  const half_t* W1H  = s_w + OW1;
  const half_t* WIHH = s_w + OWIH;
  const half_t* WHHH = s_w + OWHH;
  const half_t* W2H  = s_w + OW2;
  const unsigned obs_lds[2] = {
      (unsigned)(uintptr_t)&s_obs[w][0][0],
      (unsigned)(uintptr_t)&s_obs[w][1][0]};

  // Prologue: fill buffer 0 with the t=0 obs tile.
  issue_obs_async(obs, n0, 0, obs_lds[0], lane);

#pragma unroll 1
  for (int t = 0; t < kT; ++t) {
    WAVE_ASYNC_WAIT();                       // obs tile for step t is in LDS
    if (t + 1 < kT)                          // overlap t+1's HBM stream
      issue_obs_async(obs, n0, t + 1, obs_lds[(t + 1) & 1], lane);
    if (t + 2 < kT && lane < 16) {           // pre-pull t+2 toward L2
      int n = n0 + lane, bb = n >> 3, aa = n & 7;
      __builtin_prefetch(obs + ((size_t)bb * kT * kA +
                                (size_t)(t + 2) * kA + aa) * kOBS, 0, 1);
    }

    // last action for this lane's row (t=0 -> none -> all-zero one-hot)
    int actp = -1;
    if (t > 0) {
      int n = n0 + (lane & 15), bb = n >> 3, aa = n & 7;
      actp = act[((size_t)bb * kT + (t - 1)) * kA + aa];
    }

    // ---- x A-fragments: obs chunks from LDS, one-hot chunk from ALU
    const float* ob = &s_obs[w][t & 1][0];
    const v16h xa0 = frag_from_obs(ob, lane, 0);
    const v16h xa1 = frag_from_obs(ob, lane, 32);
    const v16h xa2 = frag_onehot(lane, aa_r, actp);

    // ---- fc1: x1 = relu(x @ W1^T + b1): 4 N-tiles x 3 K-chunks
#pragma unroll
    for (int nt = 0; nt < 4; ++nt) {
      v8f c = splat8(s_b[OB1 + nt * 16 + colc]);
      c = wmma_f16(xa0, ld_fragB(W1H, kINP, lane, nt * 16, 0), c);
      c = wmma_f16(xa1, ld_fragB(W1H, kINP, lane, nt * 16, 32), c);
      c = wmma_f16(xa2, ld_fragB(W1H, kINP, lane, nt * 16, 64), c);
      int cc = nt * 16 + colc;
#pragma unroll
      for (int v = 0; v < 8; ++v) {
        float xv = c[v] > 0.0f ? c[v] : 0.0f;
        x1b[(rbase + v) * kH + cc] = (half_t)xv;
      }
    }
    WAVE_LDS_SYNC();

    // ---- GRU step: gi = x1@W_ih^T + b_ih; gh = h@W_hh^T + b_hh; combine
    half_t* hprev = hbb + (t & 1) * (16 * kH);
    half_t* hnext = hbb + ((t + 1) & 1) * (16 * kH);
    const v16h x1a0 = ld_fragA(x1b, kH, lane, 0);
    const v16h x1a1 = ld_fragA(x1b, kH, lane, 32);
    const v16h ha0  = ld_fragA(hprev, kH, lane, 0);
    const v16h ha1  = ld_fragA(hprev, kH, lane, 32);
#pragma unroll 1
    for (int jt = 0; jt < 4; ++jt) {
      const int cr = jt * 16, cz = 64 + jt * 16, cn = 128 + jt * 16;
      v8f ir = splat8(s_b[OBIH + cr + colc]);
      v8f iz = splat8(s_b[OBIH + cz + colc]);
      v8f in_ = splat8(s_b[OBIH + cn + colc]);
      v8f hr = splat8(s_b[OBHH + cr + colc]);
      v8f hz = splat8(s_b[OBHH + cz + colc]);
      v8f hn = splat8(s_b[OBHH + cn + colc]);
      ir = wmma_f16(x1a0, ld_fragB(WIHH, kH, lane, cr, 0), ir);
      ir = wmma_f16(x1a1, ld_fragB(WIHH, kH, lane, cr, 32), ir);
      iz = wmma_f16(x1a0, ld_fragB(WIHH, kH, lane, cz, 0), iz);
      iz = wmma_f16(x1a1, ld_fragB(WIHH, kH, lane, cz, 32), iz);
      in_ = wmma_f16(x1a0, ld_fragB(WIHH, kH, lane, cn, 0), in_);
      in_ = wmma_f16(x1a1, ld_fragB(WIHH, kH, lane, cn, 32), in_);
      hr = wmma_f16(ha0, ld_fragB(WHHH, kH, lane, cr, 0), hr);
      hr = wmma_f16(ha1, ld_fragB(WHHH, kH, lane, cr, 32), hr);
      hz = wmma_f16(ha0, ld_fragB(WHHH, kH, lane, cz, 0), hz);
      hz = wmma_f16(ha1, ld_fragB(WHHH, kH, lane, cz, 32), hz);
      hn = wmma_f16(ha0, ld_fragB(WHHH, kH, lane, cn, 0), hn);
      hn = wmma_f16(ha1, ld_fragB(WHHH, kH, lane, cn, 32), hn);
#pragma unroll
      for (int v = 0; v < 8; ++v) {
        float r_ = sigmf(ir[v] + hr[v]);
        float z_ = sigmf(iz[v] + hz[v]);
        float nn = tanhf(in_[v] + r_ * hn[v]);
        float hp = (float)hprev[(rbase + v) * kH + cr + colc];
        float hnew = (1.0f - z_) * nn + z_ * hp;
        hnext[(rbase + v) * kH + cr + colc] = (half_t)hnew;
      }
    }
    WAVE_LDS_SYNC();

    // ---- qs = h_new @ W2^T + b2 (one 16-col N-tile)
    const v16h hn0 = ld_fragA(hnext, kH, lane, 0);
    const v16h hn1 = ld_fragA(hnext, kH, lane, 32);
    v8f q = splat8(s_b[OB2 + colc]);
    q = wmma_f16(hn0, ld_fragB(W2H, kH, lane, 0, 0), q);
    q = wmma_f16(hn1, ld_fragB(W2H, kH, lane, 0, 32), q);

    // ---- nn_output [B,T,A,16]
#pragma unroll
    for (int v = 0; v < 8; ++v) {
      int m = rbase + v, n = n0 + m, bb = n >> 3, aa = n & 7;
      out[(size_t)bb * (kT * kA * kACT) + (size_t)t * (kA * kACT) +
          aa * kACT + colc] = q[v];
    }
    // ---- max / argmax over 16 cols (xor tree inside 16-lane halves,
    //      first-index tie-break to match jnp.argmax)
#pragma unroll
    for (int v = 0; v < 8; ++v) {
      float val = q[v];
      int idx = colc;
#pragma unroll
      for (int mset = 8; mset >= 1; mset >>= 1) {
        float ov = __shfl_xor(val, mset);
        int oi = __shfl_xor(idx, mset);
        if (ov > val || (ov == val && oi < idx)) { val = ov; idx = oi; }
      }
      if (colc == 0) {  // lanes 0 and 16 hold rows v and v+8
        int m = rbase + v, n = n0 + m, bb = n >> 3, aa = n & 7;
        size_t pos = (size_t)bb * (kT * kA) + (size_t)t * kA + aa;
        out[MV_OFF + pos] = val;
        ((int*)out)[MA_OFF + pos] = idx;  // argmax is int32 in the reference
      }
    }
  }
}

// ------------------------------ launcher -----------------------------------
extern "C" void kernel_launch(void* const* d_in, const int* in_sizes, int n_in,
                              void* d_out, int out_size, void* d_ws,
                              size_t ws_size, hipStream_t stream) {
  const float* obs = (const float*)d_in[0];
  const int* act = (const int*)d_in[1];
  // d_in[2] = agent_ids (identity map; one-hot built from slot index)
  const float* W1 = (const float*)d_in[3];
  const float* b1 = (const float*)d_in[4];
  const float* Wih = (const float*)d_in[5];
  const float* bih = (const float*)d_in[6];
  const float* Whh = (const float*)d_in[7];
  const float* bhh = (const float*)d_in[8];
  const float* W2 = (const float*)d_in[9];
  const float* b2 = (const float*)d_in[10];
  half_t* wh = (half_t*)d_ws;  // 31744 halves = 62 KB of scratch
  float* out = (float*)d_out;

  drqn_cvt_weights<<<(WTOT + 255) / 256, 256, 0, stream>>>(W1, Wih, Whh, W2, wh);

  const int tiles = (kB * kA) / 16;               // 128 wave-tiles
  drqn_forward<<<tiles / WAVES, 32 * WAVES, 0, stream>>>(
      obs, act, b1, bih, bhh, b2, wh, out);
}
